// KNO_REG_GRID_1D_19224273617246
// MI455X (gfx1250) — matrix operand order repaired and checked by reading, not compiled, not run on hardware
//
#include <hip/hip_runtime.h>
#include <hip/hip_bf16.h>
#include <math.h>

typedef __attribute__((ext_vector_type(16))) _Float16 v16h;
typedef __attribute__((ext_vector_type(8)))  float    v8f;

#define NPTS  2048
#define CCH   32
#define CHUNK 256

__device__ __forceinline__ float gelu_tanh(float x) {
  // jax.nn.gelu default (approximate=True)
  const float k0 = 0.7978845608028654f;  // sqrt(2/pi)
  const float k1 = 0.044715f;
  float inner = k0 * (x + k1 * x * x * x);
  return 0.5f * x * (1.0f + tanhf(inner));
}

// ---------------------------------------------------------------------------
// fq = gelu([f_x, x_grid] @ lift_W^T + lift_b)
// ---------------------------------------------------------------------------
__global__ __launch_bounds__(256)
void kno_lift_kernel(const float* __restrict__ f_x, const float* __restrict__ x_grid,
                     const float* __restrict__ lift_W, const float* __restrict__ lift_b,
                     float* __restrict__ fq) {
  int i = blockIdx.x * blockDim.x + threadIdx.x;
  if (i >= NPTS * CCH) return;
  int row = i >> 5, c = i & 31;
  float v = f_x[row] * lift_W[c * 2 + 0] + x_grid[row] * lift_W[c * 2 + 1] + lift_b[c];
  fq[i] = gelu_tanh(v);
}

// ---------------------------------------------------------------------------
// One depth layer:
//   skip  = fq @ pw_W^T + pw_b                       (WMMA f32_16x16x32_f16)
//   integ = sig2_c * sum_q exp(-(x_n-x_q)^2/(2 l2_c)) * fq[q,c] * w[q]
//   fq'   = act(skip + integ)
// Block = 256 threads (8 waves) handles one 16-row tile; lane == channel.
// ---------------------------------------------------------------------------
__global__ __launch_bounds__(256)
void kno_depth_kernel(const float* __restrict__ fq_in, float* __restrict__ fq_out,
                      const float* __restrict__ x, const float* __restrict__ w,
                      const float* __restrict__ pw_W, const float* __restrict__ pw_b,
                      const float* __restrict__ log_ell, const float* __restrict__ log_sig,
                      int apply_act) {
  __shared__ float s_skip[16][CCH];          //  2 KB
  __shared__ float s_x[CHUNK];               //  1 KB
  __shared__ float s_g[CHUNK][CCH];          // 32 KB
  __shared__ float s_red[8][16][CCH];        // 16 KB

  const int tid  = threadIdx.x;
  const int wave = tid >> 5;
  const int lane = tid & 31;
  const int r0   = blockIdx.x * 16;

  // ---- skip GEMM on wave 0 (wave-uniform branch -> EXEC all ones) ----
  if (wave == 0) {
    const int khalf = lane >> 4;       // A: K-half;  B/D: lane group
    const int arow  = lane & 15;       // A row
    const int col   = lane & 15;       // B/D column
    const int kbase = khalf * 16;      // unused; B lane half starts at k = 8*khalf
    v16h a, b0, b1;
    #pragma unroll
    for (int i = 0; i < 8; ++i) {      // A 16x32 f16 layout (ISA 7.12.2)
      a[i]     = (_Float16)fq_in[(r0 + arow) * CCH + khalf * 8 + i];
      a[8 + i] = (_Float16)fq_in[(r0 + arow) * CCH + 16 + khalf * 8 + i];
    }
    #pragma unroll
    for (int j = 0; j < 16; ++j) {     // B[k,c] = pw_W[c,k]
      b0[j] = (_Float16)pw_W[col * CCH + ((j & 8) << 1) + khalf * 8 + (j & 7)];
      b1[j] = (_Float16)pw_W[(col + 16) * CCH + ((j & 8) << 1) + khalf * 8 + (j & 7)];
    }
    v8f c0 = {}, c1 = {};
    c0 = __builtin_amdgcn_wmma_f32_16x16x32_f16(false, a, false, b0, (short)0, c0, false, false);
    c1 = __builtin_amdgcn_wmma_f32_16x16x32_f16(false, a, false, b1, (short)0, c1, false, false);
    #pragma unroll
    for (int r = 0; r < 8; ++r) {      // D[r] = row 8*khalf+r, col
      s_skip[8 * khalf + r][col]      = c0[r] + pw_b[col];
      s_skip[8 * khalf + r][col + 16] = c1[r] + pw_b[col + 16];
    }
  }

  // ---- per-lane (== per-channel) Gaussian kernel constants ----
  const int c = lane;
  // exp(-d2/(2*l2)) = exp2(coef*d2), coef = -log2(e)/2 * exp(-2*log_ell)
  const float coef = -0.72134752044448170f * expf(-2.0f * log_ell[c]);
  const float sig2 = expf(2.0f * log_sig[c]);

  float xr[16], acc[16];
  #pragma unroll
  for (int r = 0; r < 16; ++r) { xr[r] = x[r0 + r]; acc[r] = 0.0f; }

  // ---- integral: stream q in LDS chunks, waves split the chunk ----
  for (int q0 = 0; q0 < NPTS; q0 += CHUNK) {
    __syncthreads();
    s_x[tid] = x[q0 + tid];
    const float wq = w[q0 + tid];
    const float4* src = (const float4*)&fq_in[(q0 + tid) * CCH];
    float4* dst = (float4*)&s_g[tid][0];
    #pragma unroll
    for (int j = 0; j < 8; ++j) {
      float4 v = src[j];
      v.x *= wq; v.y *= wq; v.z *= wq; v.w *= wq;
      dst[j] = v;
    }
    __syncthreads();
    const int qlo = wave * 32;
    #pragma unroll 4
    for (int qq = qlo; qq < qlo + 32; ++qq) {
      const float xq = s_x[qq];            // broadcast
      const float g  = s_g[qq][lane];      // conflict-free, lane == channel
      #pragma unroll
      for (int r = 0; r < 16; ++r) {
        const float d = xr[r] - xq;
        acc[r] = fmaf(g, exp2f(coef * d * d), acc[r]);
      }
    }
  }

  // ---- reduce 8 wave-partials, add skip, activate, store ----
  __syncthreads();
  #pragma unroll
  for (int r = 0; r < 16; ++r) s_red[wave][r][lane] = acc[r] * sig2;
  __syncthreads();

  #pragma unroll
  for (int pass = 0; pass < 2; ++pass) {
    const int r  = (tid >> 5) + pass * 8;
    const int cc = lane;
    float v = s_skip[r][cc];
    #pragma unroll
    for (int wv = 0; wv < 8; ++wv) v += s_red[wv][r][cc];
    if (apply_act) v = gelu_tanh(v);
    fq_out[(r0 + r) * CCH + cc] = v;
  }
}

// ---------------------------------------------------------------------------
// Projection head: gelu(fq@W1^T+b1) -> gelu(.@W2^T+b2) -> .@W3^T+b3
// One wave per 16-row tile; 4 WMMAs chained through LDS; shfl for final dot.
// ---------------------------------------------------------------------------
__global__ __launch_bounds__(256)
void kno_proj_kernel(const float* __restrict__ fq,
                     const float* __restrict__ W1, const float* __restrict__ b1,
                     const float* __restrict__ W2, const float* __restrict__ b2,
                     const float* __restrict__ W3, const float* __restrict__ b3,
                     float* __restrict__ out) {
  __shared__ float s_h1[8][16][CCH];   // 16 KB
  __shared__ float s_h2[8][16][CCH];   // 16 KB

  const int tid  = threadIdx.x;
  const int wave = tid >> 5;
  const int lane = tid & 31;
  const int r0   = (blockIdx.x * 8 + wave) * 16;

  const int khalf = lane >> 4;
  const int arow  = lane & 15;
  const int col   = lane & 15;
  const int kbase = khalf * 16;

  // ---- layer 1 ----
  v16h a, p0, p1;
  #pragma unroll
  for (int i = 0; i < 8; ++i) {
    a[i]     = (_Float16)fq[(r0 + arow) * CCH + khalf * 8 + i];
    a[8 + i] = (_Float16)fq[(r0 + arow) * CCH + 16 + khalf * 8 + i];
  }
  #pragma unroll
  for (int j = 0; j < 16; ++j) {
    p0[j] = (_Float16)W1[col * CCH + ((j & 8) << 1) + khalf * 8 + (j & 7)];
    p1[j] = (_Float16)W1[(col + 16) * CCH + ((j & 8) << 1) + khalf * 8 + (j & 7)];
  }
  {
    v8f c0 = {}, c1 = {};
    c0 = __builtin_amdgcn_wmma_f32_16x16x32_f16(false, a, false, p0, (short)0, c0, false, false);
    c1 = __builtin_amdgcn_wmma_f32_16x16x32_f16(false, a, false, p1, (short)0, c1, false, false);
    #pragma unroll
    for (int r = 0; r < 8; ++r) {
      s_h1[wave][8 * khalf + r][col]      = gelu_tanh(c0[r] + b1[col]);
      s_h1[wave][8 * khalf + r][col + 16] = gelu_tanh(c1[r] + b1[col + 16]);
    }
  }
  __syncthreads();

  // ---- layer 2 (reload h1 from LDS in A-layout) ----
  #pragma unroll
  for (int i = 0; i < 8; ++i) {
    a[i]     = (_Float16)s_h1[wave][arow][khalf * 8 + i];
    a[8 + i] = (_Float16)s_h1[wave][arow][16 + khalf * 8 + i];
  }
  #pragma unroll
  for (int j = 0; j < 16; ++j) {
    p0[j] = (_Float16)W2[col * CCH + ((j & 8) << 1) + khalf * 8 + (j & 7)];
    p1[j] = (_Float16)W2[(col + 16) * CCH + ((j & 8) << 1) + khalf * 8 + (j & 7)];
  }
  {
    v8f c0 = {}, c1 = {};
    c0 = __builtin_amdgcn_wmma_f32_16x16x32_f16(false, a, false, p0, (short)0, c0, false, false);
    c1 = __builtin_amdgcn_wmma_f32_16x16x32_f16(false, a, false, p1, (short)0, c1, false, false);
    #pragma unroll
    for (int r = 0; r < 8; ++r) {
      s_h2[wave][8 * khalf + r][col]      = gelu_tanh(c0[r] + b2[col]);
      s_h2[wave][8 * khalf + r][col + 16] = gelu_tanh(c1[r] + b2[col + 16]);
    }
  }
  __syncthreads();

  // ---- layer 3: 32 -> 1 dot, split across the two lane halves ----
  float partial = 0.0f;
  #pragma unroll
  for (int j = 0; j < 16; ++j)
    partial += s_h2[wave][arow][khalf * 16 + j] * W3[khalf * 16 + j];
  partial += __shfl_down(partial, 16);
  if (lane < 16) out[r0 + lane] = partial + b3[0];
}

// ---------------------------------------------------------------------------
extern "C" void kernel_launch(void* const* d_in, const int* in_sizes, int n_in,
                              void* d_out, int out_size, void* d_ws, size_t ws_size,
                              hipStream_t stream) {
  const float* f_x     = (const float*)d_in[0];
  const float* x_grid  = (const float*)d_in[1];
  const float* q_w     = (const float*)d_in[2];
  const float* lift_W  = (const float*)d_in[3];
  const float* lift_b  = (const float*)d_in[4];
  const float* pw_W    = (const float*)d_in[5];
  const float* pw_b    = (const float*)d_in[6];
  const float* log_ell = (const float*)d_in[7];
  const float* log_sig = (const float*)d_in[8];
  const float* p1W     = (const float*)d_in[9];
  const float* p1b     = (const float*)d_in[10];
  const float* p2W     = (const float*)d_in[11];
  const float* p2b     = (const float*)d_in[12];
  const float* p3W     = (const float*)d_in[13];
  const float* p3b     = (const float*)d_in[14];
  float* out = (float*)d_out;

  float* fqA = (float*)d_ws;             // 2048*32 f32
  float* fqB = fqA + NPTS * CCH;         // 2048*32 f32 (ping-pong)

  kno_lift_kernel<<<(NPTS * CCH + 255) / 256, 256, 0, stream>>>(
      f_x, x_grid, lift_W, lift_b, fqA);

  kno_depth_kernel<<<NPTS / 16, 256, 0, stream>>>(
      fqA, fqB, x_grid, q_w,
      pw_W + 0 * CCH * CCH, pw_b + 0 * CCH, log_ell + 0 * CCH, log_sig + 0 * CCH, 1);
  kno_depth_kernel<<<NPTS / 16, 256, 0, stream>>>(
      fqB, fqA, x_grid, q_w,
      pw_W + 1 * CCH * CCH, pw_b + 1 * CCH, log_ell + 1 * CCH, log_sig + 1 * CCH, 1);
  kno_depth_kernel<<<NPTS / 16, 256, 0, stream>>>(
      fqA, fqB, x_grid, q_w,
      pw_W + 2 * CCH * CCH, pw_b + 2 * CCH, log_ell + 2 * CCH, log_sig + 2 * CCH, 0);

  kno_proj_kernel<<<NPTS / 128, 256, 0, stream>>>(
      fqB, p1W, p1b, p2W, p2b, p3W, p3b, out);
}
